// MultiTimeframeInformer_58591943852562
// MI455X (gfx1250) — compile-verified, matrix operand order checked by reading, not run on hardware
//
#include <hip/hip_runtime.h>
#include <hip/hip_bf16.h>

// ---------------- model constants ----------------
#define BS    16
#define SEQ   512
#define PRED  24
#define DM    512
#define NH    8
#define HD    64
#define BH    (BS*NH)      // 128
#define TOK   (BS*SEQ)     // 8192
#define DTOK  (BS*PRED)    // 384
#define FFNH  2048
#define SCALE 0.125f       // 1/sqrt(64)

typedef __attribute__((ext_vector_type(16))) __bf16 v16bf;
typedef __attribute__((ext_vector_type(8)))  float  v8f;

__device__ __forceinline__ unsigned pack_bf16_pair(float lo, float hi)
{
    union { __bf16 h[2]; unsigned u; } p;
    p.h[0] = (__bf16)lo;
    p.h[1] = (__bf16)hi;
    return p.u;   // -> v_cvt_pk_bf16_f32
}

// =====================================================================
// WMMA GEMM:  C[M,N] = act( A[M,K] @ B[K,N] + bias )
// REQUIREMENTS (all call sites satisfy them): M%32==0, N%64==0, K%32==0,
// 16B-aligned rows (lda/ldb/ldc multiples of 4).
//
// Block = 256 threads = 8 waves, macro-tile 32(M) x 64(N); waves 2(M)x4(N).
// Per K-step(32): cooperative global->LDS stage (float4 loads -> bf16).
// A tile stored [m][k]; B tile stored TRANSPOSED [n][k] so that BOTH
// fragments are two contiguous 16B runs per lane -> ds_load_b128.
// B stage writes packed (k,k+1) bf16 pairs as u32 -> ds_store_b32.
// Rows padded to 48 bf16 (96B, 16B-aligned).
// =====================================================================
#define LDP 48
__global__ __launch_bounds__(256)
void gemm_wmma_kernel(const float* __restrict__ A, int lda,
                      const float* __restrict__ Bm, int ldb,
                      float* __restrict__ C, int ldc,
                      int M, int N, int K,
                      const float* __restrict__ bias, int relu)
{
    __shared__ __bf16 As [32][LDP];   // [m][k]
    __shared__ __bf16 BsT[64][LDP];   // [n][k]  (transposed B tile)

    const int tileM = blockIdx.x * 32;
    const int tileN = blockIdx.y * 64;
    const int tid   = threadIdx.x;
    const int lane  = tid & 31;
    const int wid   = tid >> 5;      // 0..7
    const int wm    = wid & 1;       // M sub-tile
    const int wn    = wid >> 1;      // N sub-tile
    const int lm    = lane & 15;     // M (A) or N (B/C) index within 16
    const int hi    = lane >> 4;     // K-half selector

    // A staging map: 4 consecutive k per thread
    const int arow = tid >> 3;              // 0..31 (m)
    const int acol = (tid & 7) * 4;         // 0..28 (k)
    // B staging map: two k-rows x 4 n per thread (writes (k,k+1) u32 pairs)
    const int bk   = (tid >> 4) * 2;        // 0..30 (k, even)
    const int bn0  = (tid & 15) * 4;        // 0..60 (n)

    const float* Abase = A  + (long)(tileM + arow) * lda + acol;
    const float* Bbase = Bm + (long)bk * ldb + tileN + bn0;

    v8f acc = {};
    for (int kk = 0; kk < K; kk += 32) {
        if (kk + 32 < K) {
            __builtin_prefetch(Abase + kk + 32, 0, 1);              // global_prefetch_b8
            __builtin_prefetch(Bbase + (long)(kk + 32) * ldb, 0, 1);
        }
        // ---- stage A (32x32 f32 -> bf16), one float4 per thread ----
        {
            const float4 v = *(const float4*)(Abase + kk);
            As[arow][acol + 0] = (__bf16)v.x;
            As[arow][acol + 1] = (__bf16)v.y;
            As[arow][acol + 2] = (__bf16)v.z;
            As[arow][acol + 3] = (__bf16)v.w;
        }
        // ---- stage B transposed (32x64 f32 -> bf16 [n][k]) ----
        // packed (k,k+1) pairs -> single ds_store_b32 each
        {
            const float4 r0 = *(const float4*)(Bbase + (long)kk * ldb);
            const float4 r1 = *(const float4*)(Bbase + (long)(kk + 1) * ldb);
            *(unsigned*)&BsT[bn0 + 0][bk] = pack_bf16_pair(r0.x, r1.x);
            *(unsigned*)&BsT[bn0 + 1][bk] = pack_bf16_pair(r0.y, r1.y);
            *(unsigned*)&BsT[bn0 + 2][bk] = pack_bf16_pair(r0.z, r1.z);
            *(unsigned*)&BsT[bn0 + 3][bk] = pack_bf16_pair(r0.w, r1.w);
        }
        __syncthreads();

        // ---- fragments: two contiguous 16B runs each -> ds_load_b128 ----
        // K map: lanes 0-15 hold K 0..7 (elems 0-7) & 16..23 (elems 8-15);
        //        lanes 16-31 hold K 8..15 & 24..31.
        v16bf af, bfv;
        const int am = wm * 16 + lm;
        const int bn = wn * 16 + lm;
        const int kb = hi * 8;
#pragma unroll
        for (int j = 0; j < 8; ++j) {
            af[j]      = As [am][kb + j];
            af[j + 8]  = As [am][kb + 16 + j];
            bfv[j]     = BsT[bn][kb + j];
            bfv[j + 8] = BsT[bn][kb + 16 + j];
        }
        acc = __builtin_amdgcn_wmma_f32_16x16x32_bf16(
                  false, af, false, bfv, (short)0, acc, false, false);
        __syncthreads();
    }

    // ---- epilogue: bias + optional ReLU, branch-free store ----
    const int n = tileN + wn * 16 + lm;
    const float bv = bias ? bias[n] : 0.f;
    float* Crow = C + (long)(tileM + wm * 16 + hi * 8) * ldc + n;
#pragma unroll
    for (int v = 0; v < 8; ++v) {       // C layout: VGPR v -> M = v (+8 for hi)
        float r = acc[v] + bv;
        if (relu) r = fmaxf(r, 0.f);
        Crow[(long)v * ldc] = r;
    }
    (void)M; (void)N;
}

// =====================================================================
// ProbSparse pass 1: streaming softmax stats -> KL(p || uniform) per row.
// kl = log(k_len) + T/Z - log(Z), with online rescaled (m, Z, T).
// One wave32 per (bh, q).
// =====================================================================
__global__ __launch_bounds__(32)
void probstats_kernel(const float* __restrict__ Q, long sQb,
                      const float* __restrict__ Km, long sKb,
                      float* __restrict__ kl, int q_len, int k_len)
{
    const int q = blockIdx.x;
    const int z = blockIdx.z;
    const int b = z / NH, h = z % NH;
    const int lane = threadIdx.x;

    const float* qrow  = Q  + (long)b * sQb + (long)q * DM + h * HD;
    const float* kbase = Km + (long)b * sKb + h * HD;

    float m = -3.4e38f, Z = 0.f, T = 0.f;
    for (int k = lane; k < k_len; k += 32) {
        const float* krow = kbase + (long)k * DM;
        float s = 0.f;
#pragma unroll 8
        for (int d = 0; d < HD; ++d) s += qrow[d] * krow[d];
        s *= SCALE;
        float e;
        if (s > m) {
            const float dd = m - s;
            const float r  = __expf(dd);
            T = r * (T + dd * Z);
            Z = r * Z;
            m = s;
            e = 1.f;
        } else {
            e = __expf(s - m);
        }
        Z += e;
        T += e * (s - m);
    }
    for (int off = 16; off; off >>= 1) {
        const float m2 = __shfl_xor(m, off, 32);
        const float Z2 = __shfl_xor(Z, off, 32);
        const float T2 = __shfl_xor(T, off, 32);
        if (Z2 > 0.f) {
            if (Z > 0.f) {
                const float mn = fmaxf(m, m2);
                const float d1 = m - mn, d2 = m2 - mn;
                const float r1 = __expf(d1), r2 = __expf(d2);
                T = r1 * (T + d1 * Z) + r2 * (T2 + d2 * Z2);
                Z = r1 * Z + r2 * Z2;
                m = mn;
            } else { m = m2; Z = Z2; T = T2; }
        }
    }
    if (lane == 0)
        kl[(long)z * q_len + q] = __logf((float)k_len) + T / Z - __logf(Z);
}

// =====================================================================
// Top-u selection by iterative wave argmax over an LDS copy of kl row.
// One wave32 per bh.  idxout stride = 32.
// =====================================================================
__global__ __launch_bounds__(32)
void topk_kernel(const float* __restrict__ kl, int* __restrict__ idxout,
                 int n, int u)
{
    __shared__ float sv[SEQ];
    const int bh = blockIdx.x;
    const int lane = threadIdx.x;
    for (int i = lane; i < n; i += 32) sv[i] = kl[(long)bh * n + i];
    __syncthreads();
    for (int t = 0; t < u; ++t) {
        float best = -3.4e38f; int bi = 0x7fffffff;
        for (int i = lane; i < n; i += 32) {
            const float v = sv[i];
            if (v > best) { best = v; bi = i; }
        }
        for (int off = 16; off; off >>= 1) {
            const float b2 = __shfl_xor(best, off, 32);
            const int   i2 = __shfl_xor(bi,   off, 32);
            if (b2 > best || (b2 == best && i2 < bi)) { best = b2; bi = i2; }
        }
        if (lane == 0) idxout[bh * 32 + t] = (bi == 0x7fffffff) ? 0 : bi;
        if (bi < n) sv[bi] = -3.4e38f;
        __syncthreads();
    }
}

// =====================================================================
// Sparse attention output for the selected query rows only.
// One wave32 per (u, bh).  Out must be pre-zeroed (scatter semantics).
// =====================================================================
__global__ __launch_bounds__(32)
void sparse_attn_kernel(const float* __restrict__ Q, long sQb,
                        const float* __restrict__ KV, long sKb,
                        const int* __restrict__ idx,
                        float* __restrict__ Out, long sOb,
                        int k_len)
{
    __shared__ float srow[SEQ];
    const int uu = blockIdx.x;
    const int z  = blockIdx.z;
    const int b  = z / NH, h = z % NH;
    const int q  = idx[z * 32 + uu];
    const int lane = threadIdx.x;

    const float* qrow  = Q  + (long)b * sQb + (long)q * DM + h * HD;
    const float* kbase = KV + (long)b * sKb + h * HD;

    float m = -3.4e38f;
    for (int k = lane; k < k_len; k += 32) {
        const float* krow = kbase + (long)k * DM;
        float s = 0.f;
#pragma unroll 8
        for (int d = 0; d < HD; ++d) s += qrow[d] * krow[d];
        s *= SCALE;
        srow[k] = s;
        m = fmaxf(m, s);
    }
    for (int off = 16; off; off >>= 1) m = fmaxf(m, __shfl_xor(m, off, 32));
    __syncthreads();
    float Z = 0.f;
    for (int k = lane; k < k_len; k += 32) {
        const float e = __expf(srow[k] - m);
        srow[k] = e;
        Z += e;
    }
    for (int off = 16; off; off >>= 1) Z += __shfl_xor(Z, off, 32);
    __syncthreads();
    const float inv = 1.f / Z;
    const int d0 = lane, d1 = lane + 32;
    float a0 = 0.f, a1 = 0.f;
    for (int k = 0; k < k_len; ++k) {
        const float p = srow[k] * inv;
        const float* vrow = kbase + (long)k * DM;
        a0 += p * vrow[d0];
        a1 += p * vrow[d1];
    }
    float* orow = Out + (long)b * sOb + (long)q * DM + h * HD;
    orow[d0] = a0;
    orow[d1] = a1;
}

// =====================================================================
// Flash-style full attention (streaming softmax, no score tensor).
// One wave32 per (q, bh); 32-key chunks staged through LDS.
// Optional causal mask (score = -1e9 for k > q, matching reference).
// =====================================================================
__global__ __launch_bounds__(32)
void full_attn_kernel(const float* __restrict__ Q, long sQb,
                      const float* __restrict__ KV, long sKb,
                      float* __restrict__ Out, long sOb,
                      int k_len, int causal)
{
    __shared__ float schunk[32];
    const int q = blockIdx.x;
    const int z = blockIdx.z;
    const int b = z / NH, h = z % NH;
    const int lane = threadIdx.x;

    const float* qrow  = Q  + (long)b * sQb + (long)q * DM + h * HD;
    const float* kbase = KV + (long)b * sKb + h * HD;

    float m = -3.4e38f, Z = 0.f, a0 = 0.f, a1 = 0.f;
    const int d0 = lane, d1 = lane + 32;

    for (int k0 = 0; k0 < k_len; k0 += 32) {
        const int k = k0 + lane;
        float s = -3.4e38f;
        if (k < k_len) {
            const float* krow = kbase + (long)k * DM;
            float t = 0.f;
#pragma unroll 8
            for (int d = 0; d < HD; ++d) t += qrow[d] * krow[d];
            s = t * SCALE;
            if (causal && k > q) s = -1e9f;
        }
        float cm = s;
        for (int off = 16; off; off >>= 1) cm = fmaxf(cm, __shfl_xor(cm, off, 32));
        const float mn = fmaxf(m, cm);
        const float r  = __expf(m - mn);
        Z *= r; a0 *= r; a1 *= r; m = mn;
        schunk[lane] = (k < k_len) ? __expf(s - m) : 0.f;
        __syncthreads();
        const int lim = (k_len - k0 < 32) ? (k_len - k0) : 32;
        for (int j = 0; j < lim; ++j) {
            const float p = schunk[j];
            const float* vrow = kbase + (long)(k0 + j) * DM;
            Z  += p;
            a0 += p * vrow[d0];
            a1 += p * vrow[d1];
        }
        __syncthreads();
    }
    const float inv = 1.f / Z;
    float* orow = Out + (long)b * sOb + (long)q * DM + h * HD;
    orow[d0] = a0 * inv;
    orow[d1] = a1 * inv;
}

// =====================================================================
// y = LayerNorm(x + r) * g + b   (population variance, eps 1e-5)
// =====================================================================
__global__ __launch_bounds__(256)
void add_ln_kernel(const float* __restrict__ X, const float* __restrict__ R,
                   float* __restrict__ Y,
                   const float* __restrict__ g, const float* __restrict__ bta)
{
    __shared__ float row[DM];
    __shared__ float red[256];
    const long t = blockIdx.x;
    const int tid = threadIdx.x;
    float s = 0.f;
    for (int d = tid; d < DM; d += 256) {
        const float v = X[t * DM + d] + R[t * DM + d];
        row[d] = v;
        s += v;
    }
    red[tid] = s; __syncthreads();
    for (int off = 128; off; off >>= 1) {
        if (tid < off) red[tid] += red[tid + off];
        __syncthreads();
    }
    const float mean = red[0] * (1.f / DM);
    __syncthreads();
    float s2 = 0.f;
    for (int d = tid; d < DM; d += 256) {
        const float v = row[d] - mean;
        s2 += v * v;
    }
    red[tid] = s2; __syncthreads();
    for (int off = 128; off; off >>= 1) {
        if (tid < off) red[tid] += red[tid + off];
        __syncthreads();
    }
    const float inv = rsqrtf(red[0] * (1.f / DM) + 1e-5f);
    for (int d = tid; d < DM; d += 256)
        Y[t * DM + d] = g[d] * (row[d] - mean) * inv + bta[d];
}

// =====================================================================
// time embedding: concat(hour, wd, mo) @ linW + linB   (384 -> 512)
// =====================================================================
__global__ __launch_bounds__(256)
void time_embed_kernel(float* __restrict__ out,
                       const int* __restrict__ hrs, const int* __restrict__ wds,
                       const int* __restrict__ mos,
                       const float* __restrict__ hourT, const float* __restrict__ wdT,
                       const float* __restrict__ moT,
                       const float* __restrict__ lw, const float* __restrict__ lb)
{
    const long t = blockIdx.x;
    const int tid = threadIdx.x;
    const float* eh = hourT + hrs[t] * 128;
    const float* ew = wdT   + wds[t] * 128;
    const float* em = moT   + mos[t] * 128;
    for (int d = tid; d < DM; d += 256) {
        float s = lb[d];
        for (int j = 0; j < 128; ++j) s += eh[j] * lw[(long)j * DM + d];
        for (int j = 0; j < 128; ++j) s += ew[j] * lw[(long)(128 + j) * DM + d];
        for (int j = 0; j < 128; ++j) s += em[j] * lw[(long)(256 + j) * DM + d];
        out[t * DM + d] = s;
    }
}

__device__ __forceinline__ float pe_val(int pos, int d)
{
    const int i = d >> 1;
    const float freq = __expf((float)(2 * i) * (-9.210340371976184f / 512.0f));
    const float x = (float)pos * freq;
    return (d & 1) ? __cosf(x) : __sinf(x);
}

// x = src @ W + b + PE + te (+ tf)
__global__ __launch_bounds__(256)
void proj_embed_kernel(const float* __restrict__ src, int din,
                       const float* __restrict__ w, const float* __restrict__ b,
                       const float* __restrict__ te, const float* __restrict__ tf,
                       float* __restrict__ out)
{
    const long t = blockIdx.x;
    const int l = (int)(t % SEQ);
    const int tid = threadIdx.x;
    const float* srow = src + t * din;
    for (int d = tid; d < DM; d += 256) {
        float s = b[d];
        for (int j = 0; j < din; ++j) s += srow[j] * w[(long)j * DM + d];
        s += pe_val(l, d) + te[t * DM + d];
        if (tf) s += tf[d];
        out[t * DM + d] = s;
    }
}

// memory = concat([x1, xd], axis=1)  -> [B, 2L, D]
__global__ void concat_kernel(const float* __restrict__ x1,
                              const float* __restrict__ xd,
                              float* __restrict__ mem)
{
    const long i = (long)blockIdx.x * 256 + threadIdx.x;
    const long tot = (long)BS * 2 * SEQ * DM;
    if (i >= tot) return;
    const long per = (long)2 * SEQ * DM;
    const long half = (long)SEQ * DM;
    const long b = i / per, r = i % per;
    mem[i] = (r < half) ? x1[b * half + r] : xd[b * half + (r - half)];
}

// dec[b,t,:] = enc[b, L-1, :] + PE[t] + te_dec[b,t,:]
__global__ __launch_bounds__(256)
void dec_init_kernel(const float* __restrict__ enc, const float* __restrict__ te,
                     float* __restrict__ dec)
{
    const long t = blockIdx.x;           // over B*PRED
    const int b = (int)(t / PRED), p = (int)(t % PRED);
    const int tid = threadIdx.x;
    const float* last = enc + ((long)b * SEQ + (SEQ - 1)) * DM;
    for (int d = tid; d < DM; d += 256)
        dec[t * DM + d] = last[d] + pe_val(p, d) + te[t * DM + d];
}

// attention pooling over time + final regressor -> out[b]
__global__ __launch_bounds__(256)
void pool_regress_kernel(const float* __restrict__ dec,
                         const float* __restrict__ wp, const float* __restrict__ bp,
                         const float* __restrict__ wr, const float* __restrict__ br,
                         float* __restrict__ out)
{
    __shared__ float logit[PRED];
    __shared__ float red[256];
    const int b = blockIdx.x;
    const int tid = threadIdx.x;
    if (tid < PRED) {
        const float* row = dec + ((long)b * PRED + tid) * DM;
        float s = bp[0];
        for (int d = 0; d < DM; ++d) s += row[d] * wp[d];
        logit[tid] = s;
    }
    __syncthreads();
    if (tid == 0) {
        float m = -3.4e38f;
        for (int i = 0; i < PRED; ++i) m = fmaxf(m, logit[i]);
        float Z = 0.f;
        for (int i = 0; i < PRED; ++i) { logit[i] = __expf(logit[i] - m); Z += logit[i]; }
        const float inv = 1.f / Z;
        for (int i = 0; i < PRED; ++i) logit[i] *= inv;
    }
    __syncthreads();
    float acc = 0.f;
    for (int d = tid; d < DM; d += 256) {
        float pooled = 0.f;
        for (int i = 0; i < PRED; ++i)
            pooled += dec[((long)b * PRED + i) * DM + d] * logit[i];
        acc += pooled * wr[d];
    }
    red[tid] = acc; __syncthreads();
    for (int off = 128; off; off >>= 1) {
        if (tid < off) red[tid] += red[tid + off];
        __syncthreads();
    }
    if (tid == 0) out[b] = red[0] + br[0];
}

// =====================================================================
// Host orchestration
// =====================================================================
extern "C" void kernel_launch(void* const* d_in, const int* in_sizes, int n_in,
                              void* d_out, int out_size, void* d_ws, size_t ws_size,
                              hipStream_t stream)
{
    (void)in_sizes; (void)n_in; (void)out_size; (void)ws_size;

    // ---- data inputs (setup_inputs order) ----
    const float* src4  = (const float*)d_in[0];
    const float* src1  = (const float*)d_in[1];
    const float* srcd  = (const float*)d_in[2];
    const int* h_enc   = (const int*)d_in[3];
    const int* w_enc   = (const int*)d_in[4];
    const int* m_enc   = (const int*)d_in[5];
    const int* h_dec   = (const int*)d_in[6];
    const int* w_dec   = (const int*)d_in[7];
    const int* m_dec   = (const int*)d_in[8];

    // ---- params: jax tree_leaves order (dict keys sorted) after index 9 ----
    auto P = [&](int i) -> const float* { return (const float*)d_in[9 + i]; };
    // 0:attn_pool.b 1:attn_pool.w
    // 2..21:   dec[2]  x {ffn1.b, ffn1.w, ffn2.b, ffn2.w, ln1.b, ln1.g, ln2.b, ln2.g, ln3.b, ln3.g}
    // 22..45:  enc1[3] x {ffn1.b, ffn1.w, ffn2.b, ffn2.w, ln1.b, ln1.g, ln2.b, ln2.g}
    // 46..69:  enc4[3] x same     70..93: encd[3] x same
    // 94:norm_cross.b 95:norm_cross.g
    // 96:proj1.b 97:proj1.w 98:proj4.b 99:proj4.w 100:projd.b 101:projd.w
    // 102:regressor.b 103:regressor.w 104:tf_emb
    // 105:time.hour 106:time.lin.b 107:time.lin.w 108:time.mo 109:time.wd
    const int DEC_BASE = 2, ENC1_BASE = 22, ENC4_BASE = 46, ENCD_BASE = 70;

    // ---- workspace bump allocator (floats) ----
    float* ws = (float*)d_ws;
    size_t o = 0;
    auto alloc = [&](size_t n) { float* p = ws + o; o += n; return p; };
    float* te_enc  = alloc((size_t)TOK * DM);
    float* x4p     = alloc((size_t)TOK * DM);
    float* x1p     = alloc((size_t)TOK * DM);
    float* xdp     = alloc((size_t)TOK * DM);
    float* memp    = alloc((size_t)BS * 2 * SEQ * DM);
    float* tmpp    = alloc((size_t)TOK * DM);
    float* hidp    = alloc((size_t)TOK * FFNH);
    float* encp    = alloc((size_t)TOK * DM);
    float* klp     = alloc((size_t)BH * SEQ);
    int*   idxp    = (int*)alloc((size_t)BH * 32);
    float* te_dec  = alloc((size_t)DTOK * DM);
    float* decp    = alloc((size_t)DTOK * DM);
    float* dtmpp   = alloc((size_t)DTOK * DM);
    float* dhidp   = alloc((size_t)DTOK * FFNH);

    const dim3 B32(32), B256(256);

    // ---- embeddings ----
    time_embed_kernel<<<TOK, B256, 0, stream>>>(te_enc, h_enc, w_enc, m_enc,
                                                P(105), P(109), P(108), P(107), P(106));
    time_embed_kernel<<<DTOK, B256, 0, stream>>>(te_dec, h_dec, w_dec, m_dec,
                                                 P(105), P(109), P(108), P(107), P(106));
    proj_embed_kernel<<<TOK, B256, 0, stream>>>(src4, 32, P(99),  P(98),  te_enc, nullptr,     x4p);
    proj_embed_kernel<<<TOK, B256, 0, stream>>>(src1, 32, P(97),  P(96),  te_enc, P(104),      x1p);
    proj_embed_kernel<<<TOK, B256, 0, stream>>>(srcd, 16, P(101), P(100), te_enc, P(104) + DM, xdp);

    // ---- encoder stacks (ProbSparse self-attn + FFN) ----
    auto enc_stream = [&](float* x, int base) {
        for (int layer = 0; layer < 3; ++layer) {
            const int pb = base + layer * 8;
            const float *f1b = P(pb+0), *f1w = P(pb+1), *f2b = P(pb+2), *f2w = P(pb+3);
            const float *l1b = P(pb+4), *l1g = P(pb+5), *l2b = P(pb+6), *l2g = P(pb+7);

            probstats_kernel<<<dim3(SEQ, 1, BH), B32, 0, stream>>>(
                x, (long)SEQ * DM, x, (long)SEQ * DM, klp, SEQ, SEQ);
            topk_kernel<<<BH, B32, 0, stream>>>(klp, idxp, SEQ, 31);
            hipMemsetAsync(tmpp, 0, (size_t)TOK * DM * sizeof(float), stream);
            sparse_attn_kernel<<<dim3(31, 1, BH), B32, 0, stream>>>(
                x, (long)SEQ * DM, x, (long)SEQ * DM, idxp, tmpp, (long)SEQ * DM, SEQ);
            add_ln_kernel<<<TOK, B256, 0, stream>>>(x, tmpp, x, l1g, l1b);

            gemm_wmma_kernel<<<dim3(TOK/32, FFNH/64, 1), B256, 0, stream>>>(
                x, DM, f1w, FFNH, hidp, FFNH, TOK, FFNH, DM, f1b, 1);
            gemm_wmma_kernel<<<dim3(TOK/32, DM/64, 1), B256, 0, stream>>>(
                hidp, FFNH, f2w, DM, tmpp, DM, TOK, DM, FFNH, f2b, 0);
            add_ln_kernel<<<TOK, B256, 0, stream>>>(x, tmpp, x, l2g, l2b);
        }
    };
    enc_stream(x4p, ENC4_BASE);
    enc_stream(x1p, ENC1_BASE);
    enc_stream(xdp, ENCD_BASE);

    // ---- cross attention: x4 attends over concat(x1, xd) ----
    {
        const long tot = (long)BS * 2 * SEQ * DM;
        concat_kernel<<<(unsigned)((tot + 255) / 256), B256, 0, stream>>>(x1p, xdp, memp);
        full_attn_kernel<<<dim3(SEQ, 1, BH), B32, 0, stream>>>(
            x4p, (long)SEQ * DM, memp, (long)2 * SEQ * DM, tmpp, (long)SEQ * DM,
            2 * SEQ, 0);
        add_ln_kernel<<<TOK, B256, 0, stream>>>(tmpp, x4p, encp, P(95), P(94));
    }

    // ---- decoder init ----
    dec_init_kernel<<<DTOK, B256, 0, stream>>>(encp, te_dec, decp);

    // ---- decoder layers ----
    for (int l = 0; l < 2; ++l) {
        const int pb = DEC_BASE + l * 10;
        const float *f1b = P(pb+0), *f1w = P(pb+1), *f2b = P(pb+2), *f2w = P(pb+3);
        const float *l1b = P(pb+4), *l1g = P(pb+5), *l2b = P(pb+6), *l2g = P(pb+7);
        const float *l3b = P(pb+8), *l3g = P(pb+9);

        // masked self-attention (causal)
        full_attn_kernel<<<dim3(PRED, 1, BH), B32, 0, stream>>>(
            decp, (long)PRED * DM, decp, (long)PRED * DM, dtmpp, (long)PRED * DM,
            PRED, 1);
        add_ln_kernel<<<DTOK, B256, 0, stream>>>(decp, dtmpp, decp, l1g, l1b);

        // ProbSparse cross-attention into encoder memory
        probstats_kernel<<<dim3(PRED, 1, BH), B32, 0, stream>>>(
            decp, (long)PRED * DM, encp, (long)SEQ * DM, klp, PRED, SEQ);
        topk_kernel<<<BH, B32, 0, stream>>>(klp, idxp, PRED, 15);
        hipMemsetAsync(dtmpp, 0, (size_t)DTOK * DM * sizeof(float), stream);
        sparse_attn_kernel<<<dim3(15, 1, BH), B32, 0, stream>>>(
            decp, (long)PRED * DM, encp, (long)SEQ * DM, idxp, dtmpp, (long)PRED * DM,
            SEQ);
        add_ln_kernel<<<DTOK, B256, 0, stream>>>(decp, dtmpp, decp, l2g, l2b);

        // FFN
        gemm_wmma_kernel<<<dim3(DTOK/32, FFNH/64, 1), B256, 0, stream>>>(
            decp, DM, f1w, FFNH, dhidp, FFNH, DTOK, FFNH, DM, f1b, 1);
        gemm_wmma_kernel<<<dim3(DTOK/32, DM/64, 1), B256, 0, stream>>>(
            dhidp, FFNH, f2w, DM, dtmpp, DM, DTOK, DM, FFNH, f2b, 0);
        add_ln_kernel<<<DTOK, B256, 0, stream>>>(decp, dtmpp, decp, l3g, l3b);
    }

    // ---- attention pooling + regressor ----
    pool_regress_kernel<<<BS, B256, 0, stream>>>(
        decp, P(1), P(0), P(103), P(102), (float*)d_out);
}
